// attention_12919261626548
// MI455X (gfx1250) — matrix operand order repaired and checked by reading, not compiled, not run on hardware
//
#include <hip/hip_runtime.h>
#include <math.h>

typedef __attribute__((ext_vector_type(16))) _Float16 v16h;
typedef __attribute__((ext_vector_type(8)))  float    v8f;

#define BATCH 4
#define CCH   64      // C
#define CQD   8       // C/8 query/key channels
#define HW    4096    // 64*64
#define IMW   64

union V16 { v16h v; _Float16 h[16]; };
union V8  { v8f  v; float f[8]; };

// ---------------------------------------------------------------------------
// Workspace layout (float elements)
// ---------------------------------------------------------------------------
constexpr size_t SZ_SFAC = 1024;                 // 3*B*64 (padded)
constexpr size_t SZ_Q    = 3ull * BATCH * CQD * HW;   // 393216
constexpr size_t SZ_V    = 3ull * BATCH * CCH * HW;   // 3145728
constexpr size_t SZ_MS   = 3ull * BATCH * HW;         // 49152
constexpr size_t SZ_CAT  = (size_t)BATCH * 192 * HW;  // 3145728
constexpr size_t SZ_C1   = (size_t)BATCH * CCH * HW;  // 1048576

constexpr size_t OFF_SFAC = 0;
constexpr size_t OFF_Q    = OFF_SFAC + SZ_SFAC;
constexpr size_t OFF_K    = OFF_Q + SZ_Q;
constexpr size_t OFF_V    = OFF_K + SZ_Q;
constexpr size_t OFF_M    = OFF_V + SZ_V;
constexpr size_t OFF_S    = OFF_M + SZ_MS;
constexpr size_t OFF_CAT  = OFF_S + SZ_MS;
constexpr size_t OFF_C1   = OFF_CAT + SZ_CAT;

// ---------------------------------------------------------------------------
// K1: channel attention -> per-(modality,b,c) scale factor
// ---------------------------------------------------------------------------
__global__ __launch_bounds__(256) void ca_kernel(
    const float* __restrict__ xr, const float* __restrict__ xd, const float* __restrict__ xt,
    const float* __restrict__ w1r, const float* __restrict__ w2r,
    const float* __restrict__ w1d, const float* __restrict__ w2d,
    const float* __restrict__ w1t, const float* __restrict__ w2t,
    float* __restrict__ sfac)
{
  int mod = blockIdx.x, b = blockIdx.y, t = threadIdx.x;
  const float* x  = (mod == 0) ? xr  : (mod == 1) ? xd  : xt;
  const float* w1 = (mod == 0) ? w1r : (mod == 1) ? w1d : w1t;
  const float* w2 = (mod == 0) ? w2r : (mod == 1) ? w2d : w2t;

  __shared__ float pm[4][64];
  __shared__ float mm[64];
  __shared__ float hh[4];

  int c = t >> 2, seg = t & 3;
  const float* xp = x + ((size_t)b * CCH + c) * HW + seg * 1024;
  float mx = -3.4e38f;
  for (int p = 0; p < 1024; ++p) mx = fmaxf(mx, xp[p]);
  pm[seg][c] = mx;
  __syncthreads();
  if (t < 64) mm[t] = fmaxf(fmaxf(pm[0][t], pm[1][t]), fmaxf(pm[2][t], pm[3][t]));
  __syncthreads();
  if (t < 4) {
    float a = 0.f;
    for (int k = 0; k < 64; ++k) a = fmaf(w1[t * 64 + k], mm[k], a);
    hh[t] = fmaxf(a, 0.f);
  }
  __syncthreads();
  if (t < 64) {
    float z = 0.f;
    for (int r = 0; r < 4; ++r) z = fmaf(w2[t * 4 + r], hh[r], z);
    sfac[(mod * BATCH + b) * CCH + t] = 1.f / (1.f + __expf(-z));
  }
}

// ---------------------------------------------------------------------------
// K2: fused 1x1 QKV projections from channel-attention-scaled inputs
// ---------------------------------------------------------------------------
__global__ __launch_bounds__(256) void qkv_kernel(
    const float* __restrict__ xr, const float* __restrict__ xd, const float* __restrict__ xt,
    const float* __restrict__ sfac,
    const float* wq0, const float* bq0, const float* wk0, const float* bk0,
    const float* wv0, const float* bv0,
    const float* wq1, const float* bq1, const float* wk1, const float* bk1,
    const float* wv1, const float* bv1,
    const float* wq2, const float* bq2, const float* wk2, const float* bk2,
    const float* wv2, const float* bv2,
    float* __restrict__ qb, float* __restrict__ kb, float* __restrict__ vb)
{
  int mod = blockIdx.z, b = blockIdx.y;
  int i = blockIdx.x * 256 + threadIdx.x;
  const float* x  = (mod == 0) ? xr  : (mod == 1) ? xd  : xt;
  const float* wq = (mod == 0) ? wq0 : (mod == 1) ? wq1 : wq2;
  const float* bq = (mod == 0) ? bq0 : (mod == 1) ? bq1 : bq2;
  const float* wk = (mod == 0) ? wk0 : (mod == 1) ? wk1 : wk2;
  const float* bk = (mod == 0) ? bk0 : (mod == 1) ? bk1 : bk2;
  const float* wv = (mod == 0) ? wv0 : (mod == 1) ? wv1 : wv2;
  const float* bv = (mod == 0) ? bv0 : (mod == 1) ? bv1 : bv2;
  const float* sf = sfac + (mod * BATCH + b) * CCH;

  float xv[64];
  #pragma unroll
  for (int c = 0; c < 64; ++c)
    xv[c] = x[((size_t)b * CCH + c) * HW + i] * sf[c];   // sf uniform -> s_load

  #pragma unroll 2
  for (int o = 0; o < CQD; ++o) {
    float aq = bq[o], ak = bk[o];
    #pragma unroll
    for (int c = 0; c < 64; ++c) {
      aq = fmaf(wq[o * 64 + c], xv[c], aq);
      ak = fmaf(wk[o * 64 + c], xv[c], ak);
    }
    qb[((size_t)(mod * BATCH + b) * CQD + o) * HW + i] = aq;
    kb[((size_t)(mod * BATCH + b) * CQD + o) * HW + i] = ak;
  }
  #pragma unroll 1
  for (int o = 0; o < CCH; ++o) {
    float av = bv[o];
    #pragma unroll
    for (int c = 0; c < 64; ++c) av = fmaf(wv[o * 64 + c], xv[c], av);
    vb[((size_t)(mod * BATCH + b) * CCH + o) * HW + i] = av;
  }
}

// ---------------------------------------------------------------------------
// K3: per-column online softmax stats over i (softmax axis=1):
//     m[j] = max_i e[i,j];  stores RECIPROCAL sum: 1 / sum_i exp(e[i,j]-m[j])
// ---------------------------------------------------------------------------
__global__ __launch_bounds__(256) void stats_kernel(
    const float* __restrict__ qb, const float* __restrict__ kb,
    float* __restrict__ mcol, float* __restrict__ scol)
{
  const int qm[3] = {1, 2, 0};   // branch -> q modality (d, t, rgb)
  const int km[3] = {2, 0, 1};   // branch -> k modality (t, rgb, d)
  int br = blockIdx.z, b = blockIdx.y;
  int j = blockIdx.x * 256 + threadIdx.x;
  const float* q = qb + (size_t)(qm[br] * BATCH + b) * CQD * HW;
  const float* k = kb + (size_t)(km[br] * BATCH + b) * CQD * HW;

  float kr[8];
  #pragma unroll
  for (int c = 0; c < 8; ++c) kr[c] = k[(size_t)c * HW + j];

  __shared__ float qs[8][512];   // 16 KB staging of q slab
  float m = -3.4e38f, s = 0.f;
  for (int ib = 0; ib < HW; ib += 512) {
    __syncthreads();
    for (int e = threadIdx.x; e < 8 * 512; e += 256) {
      int c = e >> 9, ii = e & 511;
      qs[c][ii] = q[(size_t)c * HW + ib + ii];
    }
    __syncthreads();
    for (int ii = 0; ii < 512; ++ii) {
      float e = 0.f;
      #pragma unroll
      for (int c = 0; c < 8; ++c) e = fmaf(qs[c][ii], kr[c], e);
      float nm = fmaxf(m, e);
      s = s * __expf(m - nm) + __expf(e - nm);
      m = nm;
    }
  }
  mcol[(size_t)(br * BATCH + b) * HW + j] = m;
  scol[(size_t)(br * BATCH + b) * HW + j] = 1.0f / s;   // reciprocal: PV pass multiplies
}

// ---------------------------------------------------------------------------
// K4: streaming PV pass with WMMA (f16 inputs, f32 accumulate):
//     D[i,c] = sum_j exp(e[i,j]-m[j]) * (v[c,j]*sinv[j]);  out *= (1+gamma)
// 8 waves/block, each owns a 16(i) x 64(c) strip; j streamed in chunks of 32.
// ---------------------------------------------------------------------------
__global__ __launch_bounds__(256) void pv_kernel(
    const float* __restrict__ qb, const float* __restrict__ kb,
    const float* __restrict__ vb,
    const float* __restrict__ mcol, const float* __restrict__ scol,
    const float* __restrict__ g1, const float* __restrict__ g2,
    const float* __restrict__ g3,
    float* __restrict__ cat1)
{
  const int qm[3] = {1, 2, 0};
  const int km[3] = {2, 0, 1};
  const int vm[3] = {0, 1, 2};   // branch -> v modality (rgb, d, t)
  int br = blockIdx.z, b = blockIdx.y;
  int i0 = blockIdx.x * 128;
  int t = threadIdx.x, lane = t & 31, wav = t >> 5;
  int hi = lane >> 4, ln = lane & 15;

  const float* q  = qb + (size_t)(qm[br] * BATCH + b) * CQD * HW;
  const float* k  = kb + (size_t)(km[br] * BATCH + b) * CQD * HW;
  const float* v  = vb + (size_t)(vm[br] * BATCH + b) * CCH * HW;
  const float* mc = mcol + (size_t)(br * BATCH + b) * HW;
  const float* sc = scol + (size_t)(br * BATCH + b) * HW;

  __shared__ float qi[8][128];                      // q for this i-tile (4 KB)
  __shared__ float kch[8][32];                      // k chunk (1 KB)
  __shared__ float mch[32];
  __shared__ float sch[32];                         // holds 1/s
  __shared__ alignas(16) _Float16 pA[128][32];      // p tile, A-matrix (8 KB)
  __shared__ alignas(16) _Float16 pBt[64][32];      // v' transposed (4 KB)

  for (int e = t; e < 8 * 128; e += 256) {
    int c = e >> 7, ii = e & 127;
    qi[c][ii] = q[(size_t)c * HW + i0 + ii];
  }
  __syncthreads();

  // loop-invariant: this thread's q row values for the p-tile compute
  int iiA = t & 127;            // row of p this thread computes
  int jbaseA = (t >> 7) * 16;   // 16 contiguous columns per thread
  float qv[8];
  #pragma unroll
  for (int c = 0; c < 8; ++c) qv[c] = qi[c][iiA];

  // pBt staging map: each thread owns 8 contiguous j for one channel
  int cB = t & 63;
  int j8 = (t >> 6) * 8;

  v8f acc[4];
  {
    v8f zero = {};
    #pragma unroll
    for (int ct = 0; ct < 4; ++ct) acc[ct] = zero;
  }

  for (int j0 = 0; j0 < HW; j0 += 32) {
    // ---- stage k / m / 1/s chunk ----
    if (t < 8 * 32) {
      int c = t >> 5, jj = t & 31;
      kch[c][jj] = k[(size_t)c * HW + j0 + jj];
    }
    if (t >= 256 - 32) {
      int jj = t - (256 - 32);
      mch[jj] = mc[j0 + jj];
      sch[jj] = sc[j0 + jj];
    }
    if (t == 0 && j0 + 32 < HW) __builtin_prefetch(k + j0 + 32, 0, 0);
    __syncthreads();

    // ---- p[i,j] = exp(e - m[j]) in f16; packed 16B LDS stores ----
    V16 pa;
    #pragma unroll
    for (int u = 0; u < 16; ++u) {
      int jj = jbaseA + u;
      float e = 0.f;
      #pragma unroll
      for (int c = 0; c < 8; ++c) e = fmaf(qv[c], kch[c][jj], e);
      pa.h[u] = (_Float16)__expf(e - mch[jj]);
    }
    *(uint4*)(&pA[iiA][jbaseA])     = *(uint4*)(&pa.h[0]);
    *(uint4*)(&pA[iiA][jbaseA + 8]) = *(uint4*)(&pa.h[8]);

    // ---- v'[c,j] = v[c,j] * (1/s[j]): 2x b128 loads -> 1x b128 LDS store
    {
      const float* vp = v + (size_t)cB * HW + j0 + j8;
      float4 va = *(const float4*)(vp);
      float4 vb4 = *(const float4*)(vp + 4);
      union { uint4 u4; _Float16 h[8]; } pk;
      pk.h[0] = (_Float16)(va.x * sch[j8 + 0]);
      pk.h[1] = (_Float16)(va.y * sch[j8 + 1]);
      pk.h[2] = (_Float16)(va.z * sch[j8 + 2]);
      pk.h[3] = (_Float16)(va.w * sch[j8 + 3]);
      pk.h[4] = (_Float16)(vb4.x * sch[j8 + 4]);
      pk.h[5] = (_Float16)(vb4.y * sch[j8 + 5]);
      pk.h[6] = (_Float16)(vb4.z * sch[j8 + 6]);
      pk.h[7] = (_Float16)(vb4.w * sch[j8 + 7]);
      *(uint4*)(&pBt[cB][j8]) = pk.u4;
    }
    __syncthreads();

    // ---- WMMA: A(16x32 f16) per documented layout:
    //      lane M = lane%16; elem e -> K = e + 8*(e/8) + 8*(lane>=16)
    //      => two contiguous 16B runs: [8*hi, 8*hi+7] and [16+8*hi, 23+8*hi]
    V16 a;
    const _Float16* prow = &pA[wav * 16 + ln][0];
    *(uint4*)(&a.h[0]) = *(const uint4*)(prow + 8 * hi);
    *(uint4*)(&a.h[8]) = *(const uint4*)(prow + 16 + 8 * hi);
    #pragma unroll
    for (int ct = 0; ct < 4; ++ct) {
      // B(32x16 f16): lane N = lane%16; elem e -> K = e + 8*(e/8) + 8*(lane>=16)
      V16 bb;
      const _Float16* brow = &pBt[ct * 16 + ln][0];
      *(uint4*)(&bb.h[0]) = *(const uint4*)(brow + 8 * hi);
      *(uint4*)(&bb.h[8]) = *(const uint4*)(brow + 16 + 8 * hi);
      acc[ct] = __builtin_amdgcn_wmma_f32_16x16x32_f16(
          false, a.v, false, bb.v, (short)0, acc[ct], false, false);
    }
    __syncthreads();   // protect LDS before next chunk's staging
  }

  // ---- epilogue: C/D layout M = r + 8*(lane>=16), N = lane%16 ----
  float g = 1.f + ((br == 0) ? g1[0] : (br == 1) ? g2[0] : g3[0]);
  #pragma unroll
  for (int ct = 0; ct < 4; ++ct) {
    V8 u8; u8.v = acc[ct];
    int c = ct * 16 + ln;
    #pragma unroll
    for (int r = 0; r < 8; ++r) {
      int ii = wav * 16 + r + 8 * hi;
      cat1[((size_t)b * 192 + br * 64 + c) * HW + i0 + ii] = g * u8.f[r];
    }
  }
}

// ---------------------------------------------------------------------------
// K5: 3x3 conv (192 -> 64, SAME) + BN(eval) + ReLU
// ---------------------------------------------------------------------------
__global__ __launch_bounds__(256) void conv1_kernel(
    const float* __restrict__ xin, const float* __restrict__ w,
    const float* __restrict__ bscale, const float* __restrict__ bbias,
    const float* __restrict__ bmean, const float* __restrict__ bvar,
    float* __restrict__ outp)
{
  int b = blockIdx.y;
  int tile = blockIdx.x, tx = (tile & 3) * 16, ty = (tile >> 2) * 16;
  int t = threadIdx.x, px = t & 15, py = t >> 4;
  __shared__ float xt[16][18][18];   // 16-channel halo tile (~20.7 KB)

  float acc[64];
  #pragma unroll
  for (int o = 0; o < 64; ++o) acc[o] = 0.f;

  for (int cb = 0; cb < 192; cb += 16) {
    __syncthreads();
    for (int e = t; e < 16 * 18 * 18; e += 256) {
      int ci = e / 324, rem = e % 324, yy = rem / 18, xx = rem % 18;
      int gx = tx + xx - 1, gy = ty + yy - 1;
      float vx = 0.f;
      if (gx >= 0 && gx < IMW && gy >= 0 && gy < IMW)
        vx = xin[((size_t)b * 192 + cb + ci) * HW + gy * IMW + gx];
      xt[ci][yy][xx] = vx;
    }
    __syncthreads();
    for (int ci = 0; ci < 16; ++ci) {
      float xv[9];
      #pragma unroll
      for (int ky = 0; ky < 3; ++ky)
        #pragma unroll
        for (int kx = 0; kx < 3; ++kx)
          xv[ky * 3 + kx] = xt[ci][py + ky][px + kx];
      const float* wb = w + (size_t)(cb + ci) * 9;   // + o*192*9, uniform -> s_load
      #pragma unroll
      for (int o = 0; o < 64; ++o) {
        const float* wp = wb + (size_t)o * (192 * 9);
        float a = acc[o];
        #pragma unroll
        for (int tp = 0; tp < 9; ++tp) a = fmaf(wp[tp], xv[tp], a);
        acc[o] = a;
      }
    }
  }
  int pos = (ty + py) * IMW + tx + px;
  #pragma unroll
  for (int o = 0; o < 64; ++o) {
    float inv = bscale[o] * rsqrtf(bvar[o] + 1e-5f);
    float y = (acc[o] - bmean[o]) * inv + bbias[o];
    outp[((size_t)b * 64 + o) * HW + pos] = fmaxf(y, 0.f);
  }
}

// ---------------------------------------------------------------------------
// K6: fused concat(f5*out, f5+out) -> 3x3 conv (128 -> 64) + BN + ReLU -> d_out
// ---------------------------------------------------------------------------
__global__ __launch_bounds__(256) void conv2_kernel(
    const float* __restrict__ c1, const float* __restrict__ f5,
    const float* __restrict__ w,
    const float* __restrict__ bscale, const float* __restrict__ bbias,
    const float* __restrict__ bmean, const float* __restrict__ bvar,
    float* __restrict__ outp)
{
  int b = blockIdx.y;
  int tile = blockIdx.x, tx = (tile & 3) * 16, ty = (tile >> 2) * 16;
  int t = threadIdx.x, px = t & 15, py = t >> 4;
  __shared__ float xt[16][18][18];

  float acc[64];
  #pragma unroll
  for (int o = 0; o < 64; ++o) acc[o] = 0.f;

  for (int cb = 0; cb < 128; cb += 16) {
    __syncthreads();
    for (int e = t; e < 16 * 18 * 18; e += 256) {
      int ci = e / 324, rem = e % 324, yy = rem / 18, xx = rem % 18;
      int gx = tx + xx - 1, gy = ty + yy - 1;
      float vx = 0.f;
      if (gx >= 0 && gx < IMW && gy >= 0 && gy < IMW) {
        int cc = cb + ci;
        int src = cc & 63;
        size_t idx = ((size_t)b * 64 + src) * HW + gy * IMW + gx;
        float f = f5[idx], o1 = c1[idx];
        vx = (cc < 64) ? (f * o1) : (f + o1);
      }
      xt[ci][yy][xx] = vx;
    }
    __syncthreads();
    for (int ci = 0; ci < 16; ++ci) {
      float xv[9];
      #pragma unroll
      for (int ky = 0; ky < 3; ++ky)
        #pragma unroll
        for (int kx = 0; kx < 3; ++kx)
          xv[ky * 3 + kx] = xt[ci][py + ky][px + kx];
      const float* wb = w + (size_t)(cb + ci) * 9;
      #pragma unroll
      for (int o = 0; o < 64; ++o) {
        const float* wp = wb + (size_t)o * (128 * 9);
        float a = acc[o];
        #pragma unroll
        for (int tp = 0; tp < 9; ++tp) a = fmaf(wp[tp], xv[tp], a);
        acc[o] = a;
      }
    }
  }
  int pos = (ty + py) * IMW + tx + px;
  #pragma unroll
  for (int o = 0; o < 64; ++o) {
    float inv = bscale[o] * rsqrtf(bvar[o] + 1e-5f);
    float y = (acc[o] - bmean[o]) * inv + bbias[o];
    outp[((size_t)b * 64 + o) * HW + pos] = fmaxf(y, 0.f);
  }
}

// ---------------------------------------------------------------------------
extern "C" void kernel_launch(void* const* d_in, const int* in_sizes, int n_in,
                              void* d_out, int out_size, void* d_ws, size_t ws_size,
                              hipStream_t stream)
{
  (void)in_sizes; (void)n_in; (void)out_size; (void)ws_size;
  const float* rgb = (const float*)d_in[0];
  const float* dd  = (const float*)d_in[1];
  const float* tt  = (const float*)d_in[2];
  const float* f5  = (const float*)d_in[3];
  const float* ca_w1_r = (const float*)d_in[4];
  const float* ca_w2_r = (const float*)d_in[5];
  const float* ca_w1_d = (const float*)d_in[6];
  const float* ca_w2_d = (const float*)d_in[7];
  const float* ca_w1_t = (const float*)d_in[8];
  const float* ca_w2_t = (const float*)d_in[9];
  const float* wq_r = (const float*)d_in[10]; const float* bq_r = (const float*)d_in[11];
  const float* wk_r = (const float*)d_in[12]; const float* bk_r = (const float*)d_in[13];
  const float* wv_r = (const float*)d_in[14]; const float* bv_r = (const float*)d_in[15];
  const float* wq_d = (const float*)d_in[16]; const float* bq_d = (const float*)d_in[17];
  const float* wk_d = (const float*)d_in[18]; const float* bk_d = (const float*)d_in[19];
  const float* wv_d = (const float*)d_in[20]; const float* bv_d = (const float*)d_in[21];
  const float* wq_t = (const float*)d_in[22]; const float* bq_t = (const float*)d_in[23];
  const float* wk_t = (const float*)d_in[24]; const float* bk_t = (const float*)d_in[25];
  const float* wv_t = (const float*)d_in[26]; const float* bv_t = (const float*)d_in[27];
  const float* g1 = (const float*)d_in[28];
  const float* g2 = (const float*)d_in[29];
  const float* g3 = (const float*)d_in[30];
  const float* conv1_w = (const float*)d_in[31];
  const float* bn1s = (const float*)d_in[32]; const float* bn1b = (const float*)d_in[33];
  const float* bn1m = (const float*)d_in[34]; const float* bn1v = (const float*)d_in[35];
  const float* conv2_w = (const float*)d_in[36];
  const float* bn2s = (const float*)d_in[37]; const float* bn2b = (const float*)d_in[38];
  const float* bn2m = (const float*)d_in[39]; const float* bn2v = (const float*)d_in[40];

  float* ws   = (float*)d_ws;
  float* sfac = ws + OFF_SFAC;
  float* qbuf = ws + OFF_Q;
  float* kbuf = ws + OFF_K;
  float* vbuf = ws + OFF_V;
  float* mcol = ws + OFF_M;
  float* scol = ws + OFF_S;
  float* cat1 = ws + OFF_CAT;
  float* c1o  = ws + OFF_C1;

  ca_kernel<<<dim3(3, BATCH), 256, 0, stream>>>(
      rgb, dd, tt, ca_w1_r, ca_w2_r, ca_w1_d, ca_w2_d, ca_w1_t, ca_w2_t, sfac);

  qkv_kernel<<<dim3(HW / 256, BATCH, 3), 256, 0, stream>>>(
      rgb, dd, tt, sfac,
      wq_r, bq_r, wk_r, bk_r, wv_r, bv_r,
      wq_d, bq_d, wk_d, bk_d, wv_d, bv_d,
      wq_t, bq_t, wk_t, bk_t, wv_t, bv_t,
      qbuf, kbuf, vbuf);

  stats_kernel<<<dim3(HW / 256, BATCH, 3), 256, 0, stream>>>(
      qbuf, kbuf, mcol, scol);

  pv_kernel<<<dim3(HW / 128, BATCH, 3), 256, 0, stream>>>(
      qbuf, kbuf, vbuf, mcol, scol, g1, g2, g3, cat1);

  conv1_kernel<<<dim3(16, BATCH), 256, 0, stream>>>(
      cat1, conv1_w, bn1s, bn1b, bn1m, bn1v, c1o);

  conv2_kernel<<<dim3(16, BATCH), 256, 0, stream>>>(
      c1o, f5, conv2_w, bn2s, bn2b, bn2m, bn2v, (float*)d_out);
}
